// SparseBlock_25280177504759
// MI455X (gfx1250) — matrix operand order repaired and checked by reading, not compiled, not run on hardware
//
#include <hip/hip_runtime.h>

typedef __attribute__((ext_vector_type(16))) _Float16 v16h;
typedef __attribute__((ext_vector_type(8)))  float    v8f;

#define NPTS   (1 << 20)
#define CIN    32
#define COUT   64
#define KNB    9
#define BN_EPS 1e-4f

#define WAVES_PER_BLOCK 8
#define ROWS_PER_WAVE   16
#define ROWS_PER_BLOCK  (WAVES_PER_BLOCK * ROWS_PER_WAVE)   // 128
#define NCHUNK          (NPTS / ROWS_PER_BLOCK)             // 8192
#define GRID1           4096
#define CHUNK_ITERS     (NCHUNK / GRID1)                    // 2

#define RED1_BLOCKS     64
#define RED1_SPAN       (GRID1 / RED1_BLOCKS)               // 64

// ---------------------------------------------------------------------------
// Pass A: gathered GEMM (v_wmma_f32_16x16x32_f16) -> BN statistics ONLY.
// ---------------------------------------------------------------------------
__global__ __launch_bounds__(256) void spconv_stats_passA(
    const float* __restrict__ features,   // [N][32] fp32
    const float* __restrict__ weight,     // [9][32][64] fp32
    const int*   __restrict__ nbr,        // [N][9]
    float*       __restrict__ partials)   // [GRID1][128] (sum64, sumsq64)
{
    __shared__ __attribute__((aligned(32))) _Float16 wT[KNB * COUT * CIN]; // 36 KB
    __shared__ float redS[WAVES_PER_BLOCK][128];
    __shared__ float redQ[WAVES_PER_BLOCK][128];

    const int tid = threadIdx.x;

    // Stage + transpose + f32->f16 convert weights: wT[k][cout][cin]
    for (int e = tid; e < KNB * CIN * COUT; e += 256) {
        int k  = e / (CIN * COUT);
        int r  = e - k * (CIN * COUT);
        int ci = r / COUT;
        int co = r - ci * COUT;
        wT[(k * COUT + co) * CIN + ci] = (_Float16)weight[e];
    }
    __syncthreads();

    const int wave = tid >> 5;
    const int lane = tid & 31;
    const int mrow = lane & 15;
    const int hi   = lane >> 4;
    const int c0   = hi * 8;

    float sumAcc[4] = {0.f, 0.f, 0.f, 0.f};
    float sqAcc[4]  = {0.f, 0.f, 0.f, 0.f};

    const float4* f4 = (const float4*)features;

    for (int it = 0; it < CHUNK_ITERS; ++it) {
        const int chunk = blockIdx.x + it * GRID1;
        const int row0  = chunk * ROWS_PER_BLOCK + wave * ROWS_PER_WAVE;
        const int r     = row0 + mrow;

        // One clause of 9 index loads, held in static registers (loop below is
        // fully unrolled, so no dynamic VGPR indexing / movrels).
        int nb9[KNB];
#pragma unroll
        for (int k = 0; k < KNB; ++k)
            nb9[k] = nbr[r * KNB + k];

        v8f acc[4] = {{0.f,0.f,0.f,0.f,0.f,0.f,0.f,0.f},
                      {0.f,0.f,0.f,0.f,0.f,0.f,0.f,0.f},
                      {0.f,0.f,0.f,0.f,0.f,0.f,0.f,0.f},
                      {0.f,0.f,0.f,0.f,0.f,0.f,0.f,0.f}};

#pragma unroll
        for (int k = 0; k < KNB; ++k) {
            const int   nb   = nb9[k];
            const float m    = (nb >= 0) ? 1.0f : 0.0f;
            const int   safe = (nb >= 0) ? nb : 0;
            const int   b    = safe * 8 + (c0 >> 1);
            const float4 fa = f4[b];
            const float4 fb = f4[b + 1];
            const float4 fc = f4[b + 2];
            const float4 fd = f4[b + 3];

            v16h a;
            a[0]  = (_Float16)(fa.x * m);  a[1]  = (_Float16)(fa.y * m);
            a[2]  = (_Float16)(fa.z * m);  a[3]  = (_Float16)(fa.w * m);
            a[4]  = (_Float16)(fb.x * m);  a[5]  = (_Float16)(fb.y * m);
            a[6]  = (_Float16)(fb.z * m);  a[7]  = (_Float16)(fb.w * m);
            a[8]  = (_Float16)(fc.x * m);  a[9]  = (_Float16)(fc.y * m);
            a[10] = (_Float16)(fc.z * m);  a[11] = (_Float16)(fc.w * m);
            a[12] = (_Float16)(fd.x * m);  a[13] = (_Float16)(fd.y * m);
            a[14] = (_Float16)(fd.z * m);  a[15] = (_Float16)(fd.w * m);

#pragma unroll
            for (int nt = 0; nt < 4; ++nt) {
                const int co = nt * 16 + mrow;
                const v16h bfrag =
                    *(const v16h*)&wT[(k * COUT + co) * CIN + hi * 16];
                acc[nt] = __builtin_amdgcn_wmma_f32_16x16x32_f16(
                    false, a, false, bfrag, (short)0, acc[nt], false, false);
            }
        }

#pragma unroll
        for (int nt = 0; nt < 4; ++nt) {
            float s = 0.f, q = 0.f;
#pragma unroll
            for (int v = 0; v < 8; ++v) {
                const float x = acc[nt][v];
                s += x;
                q += x * x;
            }
            sumAcc[nt] += s;
            sqAcc[nt]  += q;
        }
    }

    // Deterministic in-block reduction.
#pragma unroll
    for (int nt = 0; nt < 4; ++nt) {
        const int idx = (nt * 16 + mrow) * 2 + hi;
        redS[wave][idx] = sumAcc[nt];
        redQ[wave][idx] = sqAcc[nt];
    }
    __syncthreads();

    if (tid < 64) {
        float s = 0.f;
        for (int w = 0; w < WAVES_PER_BLOCK; ++w)
            s += redS[w][2 * tid] + redS[w][2 * tid + 1];
        partials[blockIdx.x * 128 + tid] = s;
    } else if (tid < 128) {
        const int t = tid - 64;
        float q = 0.f;
        for (int w = 0; w < WAVES_PER_BLOCK; ++w)
            q += redQ[w][2 * t] + redQ[w][2 * t + 1];
        partials[blockIdx.x * 128 + 64 + t] = q;
    }
}

// ---------------------------------------------------------------------------
// Pass B1: first-level tree reduce of the 4096 block partials (64 ways).
// ---------------------------------------------------------------------------
__global__ __launch_bounds__(128) void bn_reduce1(
    const float* __restrict__ partials,   // [GRID1][128]
    float*       __restrict__ partial2)   // [RED1_BLOCKS][128]
{
    const int t = threadIdx.x;
    const int g = blockIdx.x;
    float a = 0.f;
    const float* p = partials + (long)g * RED1_SPAN * 128;
    for (int b = 0; b < RED1_SPAN; ++b)
        a += p[b * 128 + t];
    partial2[g * 128 + t] = a;
}

// ---------------------------------------------------------------------------
// Pass B2: final reduce -> per-channel scale/shift.
// ---------------------------------------------------------------------------
__global__ __launch_bounds__(128) void bn_reduce2(
    const float* __restrict__ partial2,
    const float* __restrict__ gamma,
    const float* __restrict__ beta,
    float*       __restrict__ params)     // [scale64, shift64]
{
    __shared__ float tot[128];
    const int t = threadIdx.x;
    float a = 0.f;
    for (int g = 0; g < RED1_BLOCKS; ++g)
        a += partial2[g * 128 + t];
    tot[t] = a;
    __syncthreads();
    if (t < 64) {
        const float invN  = 1.0f / (float)NPTS;
        const float mean  = tot[t] * invN;
        const float var   = tot[64 + t] * invN - mean * mean;
        const float scale = gamma[t] * rsqrtf(var + BN_EPS);
        params[t]      = scale;
        params[64 + t] = beta[t] - mean * scale;
    }
}

// ---------------------------------------------------------------------------
// Pass C: recompute the GEMM (inputs now L2-hot) and emit the final
// relu(x*scale + shift) directly -- out is written exactly once.
// ---------------------------------------------------------------------------
__global__ __launch_bounds__(256) void spconv_gemm_passC(
    const float* __restrict__ features,
    const float* __restrict__ weight,
    const int*   __restrict__ nbr,
    const float* __restrict__ params,     // [scale64, shift64]
    float*       __restrict__ out)        // [N][64] final
{
    __shared__ __attribute__((aligned(32))) _Float16 wT[KNB * COUT * CIN];
    __shared__ float sc[64];
    __shared__ float sh[64];

    const int tid = threadIdx.x;

    for (int e = tid; e < KNB * CIN * COUT; e += 256) {
        int k  = e / (CIN * COUT);
        int r  = e - k * (CIN * COUT);
        int ci = r / COUT;
        int co = r - ci * COUT;
        wT[(k * COUT + co) * CIN + ci] = (_Float16)weight[e];
    }
    if (tid < 64) {
        sc[tid] = params[tid];
        sh[tid] = params[64 + tid];
    }
    __syncthreads();

    const int wave = tid >> 5;
    const int lane = tid & 31;
    const int mrow = lane & 15;
    const int hi   = lane >> 4;
    const int c0   = hi * 8;

    float scv[4], shv[4];
#pragma unroll
    for (int nt = 0; nt < 4; ++nt) {
        scv[nt] = sc[nt * 16 + mrow];
        shv[nt] = sh[nt * 16 + mrow];
    }

    const float4* f4 = (const float4*)features;

    for (int it = 0; it < CHUNK_ITERS; ++it) {
        const int chunk = blockIdx.x + it * GRID1;
        const int row0  = chunk * ROWS_PER_BLOCK + wave * ROWS_PER_WAVE;
        const int r     = row0 + mrow;

        int nb9[KNB];
#pragma unroll
        for (int k = 0; k < KNB; ++k)
            nb9[k] = nbr[r * KNB + k];

        v8f acc[4] = {{0.f,0.f,0.f,0.f,0.f,0.f,0.f,0.f},
                      {0.f,0.f,0.f,0.f,0.f,0.f,0.f,0.f},
                      {0.f,0.f,0.f,0.f,0.f,0.f,0.f,0.f},
                      {0.f,0.f,0.f,0.f,0.f,0.f,0.f,0.f}};

#pragma unroll
        for (int k = 0; k < KNB; ++k) {
            const int   nb   = nb9[k];
            const float m    = (nb >= 0) ? 1.0f : 0.0f;
            const int   safe = (nb >= 0) ? nb : 0;
            const int   b    = safe * 8 + (c0 >> 1);
            const float4 fa = f4[b];
            const float4 fb = f4[b + 1];
            const float4 fc = f4[b + 2];
            const float4 fd = f4[b + 3];

            v16h a;
            a[0]  = (_Float16)(fa.x * m);  a[1]  = (_Float16)(fa.y * m);
            a[2]  = (_Float16)(fa.z * m);  a[3]  = (_Float16)(fa.w * m);
            a[4]  = (_Float16)(fb.x * m);  a[5]  = (_Float16)(fb.y * m);
            a[6]  = (_Float16)(fb.z * m);  a[7]  = (_Float16)(fb.w * m);
            a[8]  = (_Float16)(fc.x * m);  a[9]  = (_Float16)(fc.y * m);
            a[10] = (_Float16)(fc.z * m);  a[11] = (_Float16)(fc.w * m);
            a[12] = (_Float16)(fd.x * m);  a[13] = (_Float16)(fd.y * m);
            a[14] = (_Float16)(fd.z * m);  a[15] = (_Float16)(fd.w * m);

#pragma unroll
            for (int nt = 0; nt < 4; ++nt) {
                const int co = nt * 16 + mrow;
                const v16h bfrag =
                    *(const v16h*)&wT[(k * COUT + co) * CIN + hi * 16];
                acc[nt] = __builtin_amdgcn_wmma_f32_16x16x32_f16(
                    false, a, false, bfrag, (short)0, acc[nt], false, false);
            }
        }

        // Fused BN + ReLU epilogue; each element written exactly once.
#pragma unroll
        for (int nt = 0; nt < 4; ++nt) {
            const int col = nt * 16 + mrow;
#pragma unroll
            for (int v = 0; v < 8; ++v) {
                const int rr = row0 + v + hi * 8;
                const float x = fmaxf(acc[nt][v] * scv[nt] + shv[nt], 0.f);
                out[rr * COUT + col] = x;
            }
        }
    }
}

// ---------------------------------------------------------------------------
extern "C" void kernel_launch(void* const* d_in, const int* in_sizes, int n_in,
                              void* d_out, int out_size, void* d_ws, size_t ws_size,
                              hipStream_t stream) {
    const float* features = (const float*)d_in[0];
    const float* weight   = (const float*)d_in[1];
    const float* gamma    = (const float*)d_in[2];
    const float* beta     = (const float*)d_in[3];
    const int*   nbr      = (const int*)d_in[4];
    float* out = (float*)d_out;

    float* partials = (float*)d_ws;                      // 4096*128 floats
    float* partial2 = partials + GRID1 * 128;            // 64*128 floats
    float* params   = partial2 + RED1_BLOCKS * 128;      // 128 floats

    spconv_stats_passA<<<GRID1, 256, 0, stream>>>(features, weight, nbr, partials);
    bn_reduce1<<<RED1_BLOCKS, 128, 0, stream>>>(partials, partial2);
    bn_reduce2<<<1, 128, 0, stream>>>(partial2, gamma, beta, params);
    spconv_gemm_passC<<<GRID1, 256, 0, stream>>>(features, weight, nbr, params, out);
}
